// GraphUpsamplingBlock_21071109554686
// MI455X (gfx1250) — matrix operand order repaired and checked by reading, not compiled, not run on hardware
//
#include <hip/hip_runtime.h>

typedef _Float16 half_t;
typedef __attribute__((ext_vector_type(16))) _Float16 v16h;
typedef __attribute__((ext_vector_type(8)))  _Float16 v8h;
typedef __attribute__((ext_vector_type(8)))  float    v8f;
typedef __attribute__((ext_vector_type(4)))  float    f32x4;

#define N_COARSE 55552          // 248*224
#define WCg      224
#define WFg      448
#define CDIM     128
#define OUTD     256

// Workspace: f16 weight fragments, exact WMMA A-operand layout.
// One fragment = 32 lanes * 16 halves = 1024 B (512 halves).
#define FRAG_H    512
#define OFF_EMB_H 0                 // 4q * 4mt * 4ks  = 64 frags
#define OFF_WE_H  (64 * FRAG_H)    // 4q * 4mt * 2kb  = 32 frags (pre-scaled by 0.25)
#define OFF_WN_H  (96 * FRAG_H)    // 4q * 16mt * 2kb = 128 frags
// total = 224 fragments = 229376 bytes

// A-fragment (f16 16x32) K index within a 32-block for (hi = lane>>4, j = half 0..15):
// k = hi*16 + j   (contiguous order, matching the B operands)
__device__ __forceinline__ int klocal_of(int hi, int j) {
    return hi * 16 + j;
}

#define WMMA_F16(A, B, C) \
    __builtin_amdgcn_wmma_f32_16x16x32_f16(false, (A), false, (B), (short)0, (C), false, false)

// ---------------------------------------------------------------------------
// Prep: rearrange f32 weights into f16 A-fragment layout in d_ws.
// One thread builds one (fragment, lane) = 16 halves = 32 B.
// Tasks: emb 64*32=2048, we 32*32=1024, wn 128*32=4096 -> 7168 total.
// ---------------------------------------------------------------------------
__global__ void __launch_bounds__(256)
prep_weights(const float* __restrict__ W_emb,
             const float* __restrict__ W_edge,
             const float* __restrict__ W_node,
             half_t* __restrict__ wsf)
{
    int tid = blockIdx.x * 256 + threadIdx.x;
    if (tid >= 7168) return;

    v16h frag;
    half_t* dst;

    if (tid < 2048) {                       // W_emb[q] : 128x64, A = W^T (64x128)
        int f = tid >> 5, L = tid & 31;
        int ks = f & 3, mt = (f >> 2) & 3, q = f >> 4;
        int col = mt * 16 + (L & 15), hi = L >> 4;
#pragma unroll
        for (int j = 0; j < 16; ++j) {
            int k = ks * 32 + klocal_of(hi, j);
            frag[j] = (half_t)W_emb[(q * 128 + k) * 64 + col];
        }
        dst = wsf + OFF_EMB_H + f * FRAG_H + L * 16;
    } else if (tid < 3072) {                // W_edge[q] sender rows 2..65 (64x64), *0.25
        int t = tid - 2048;
        int f = t >> 5, L = t & 31;
        int kb = f & 1, mt = (f >> 1) & 3, q = f >> 3;
        int col = mt * 16 + (L & 15), hi = L >> 4;
#pragma unroll
        for (int j = 0; j < 16; ++j) {
            int k = kb * 32 + klocal_of(hi, j);
            frag[j] = (half_t)(0.25f * W_edge[(q * 130 + 2 + k) * 64 + col]);
        }
        dst = wsf + OFF_WE_H + f * FRAG_H + L * 16;
    } else {                                // W_node rows 128+64q .. +63 (64x256)
        int t = tid - 3072;
        int f = t >> 5, L = t & 31;
        int kb = f & 1, mt = (f >> 1) & 15, q = f >> 5;
        int col = mt * 16 + (L & 15), hi = L >> 4;
#pragma unroll
        for (int j = 0; j < 16; ++j) {
            int k = kb * 32 + klocal_of(hi, j);
            frag[j] = (half_t)W_node[(128 + q * 64 + k) * 256 + col];
        }
        dst = wsf + OFF_WN_H + f * FRAG_H + L * 16;
    }
    *(v16h*)dst = frag;
}

// ---------------------------------------------------------------------------
// Main: one wave = one 16-coarse-node tile; 8 waves/block; 434 blocks = 3472 tiles.
// Chain (per quadrant): D1=relu(WembT x XT) -> LDS -> D2=relu(WeT x H) -> LDS ->
// D3=relu(WnT x G) -> LDS transpose -> coalesced 256B scatter stores.
// ---------------------------------------------------------------------------
__global__ void __launch_bounds__(256)
upsample_kernel(const float* __restrict__ x,
                const half_t* __restrict__ wsf,
                float* __restrict__ out)
{
    __shared__ unsigned char smem_all[8 * 4352];
    const int lane = threadIdx.x & 31;
    const int wave = threadIdx.x >> 5;
    const int ln = lane & 15;
    const int hi = lane >> 4;
    half_t* ldsH = (half_t*)(smem_all + wave * 4352);  // 16 nodes x 72 halves
    float*  ldsO = (float*) (smem_all + wave * 4352);  // 16 nodes x 68 floats
    const int tile = blockIdx.x * 8 + wave;
    const int m0 = tile * 16;

    // X^T B-fragments (lane=node, 16 contiguous K per lane): 64B/lane loads, f32->f16.
    v16h B1[4];
    {
        const float* xr = x + (size_t)(m0 + ln) * CDIM + hi * 16;
#pragma unroll
        for (int ks = 0; ks < 4; ++ks) {
            f32x4 a0 = *(const f32x4*)(xr + ks * 32);
            f32x4 a1 = *(const f32x4*)(xr + ks * 32 + 4);
            f32x4 a2 = *(const f32x4*)(xr + ks * 32 + 8);
            f32x4 a3 = *(const f32x4*)(xr + ks * 32 + 12);
            v16h b;
#pragma unroll
            for (int i = 0; i < 4; ++i) {
                b[i]      = (half_t)a0[i];
                b[4 + i]  = (half_t)a1[i];
                b[8 + i]  = (half_t)a2[i];
                b[12 + i] = (half_t)a3[i];
            }
            B1[ks] = b;
        }
    }

    for (int q = 0; q < 4; ++q) {
        // ---- stage 1: H^T = relu(Wemb^T @ X^T), 4 Mtiles x 4 Ksteps = 16 WMMA
        const half_t* wemb_q = wsf + OFF_EMB_H + (q * 16) * FRAG_H;
#pragma unroll
        for (int mt = 0; mt < 4; ++mt) {
            v8f acc = {};
#pragma unroll
            for (int ks = 0; ks < 4; ++ks) {
                v16h A = *(const v16h*)(wemb_q + (mt * 4 + ks) * FRAG_H + lane * 16);
                acc = WMMA_F16(A, B1[ks], acc);
            }
            v8h h;
#pragma unroll
            for (int i = 0; i < 8; ++i) h[i] = (half_t)fmaxf(acc[i], 0.0f);
            *(v8h*)(ldsH + ln * 72 + mt * 16 + hi * 8) = h;   // node-major, pad 72
        }
        asm volatile("s_wait_dscnt 0" ::: "memory");

        // ---- stage 2: G^T = relu(We'^T @ H^T), 4x2 = 8 WMMA (0.25 pre-folded)
        v16h B2[2];
        B2[0] = *(const v16h*)(ldsH + ln * 72 + hi * 16);
        B2[1] = *(const v16h*)(ldsH + ln * 72 + 32 + hi * 16);
        asm volatile("" ::: "memory");   // keep loads before the G stores below
        const half_t* we_q = wsf + OFF_WE_H + (q * 8) * FRAG_H;
#pragma unroll
        for (int mt = 0; mt < 4; ++mt) {
            v8f acc = {};
#pragma unroll
            for (int kb = 0; kb < 2; ++kb) {
                v16h A = *(const v16h*)(we_q + (mt * 2 + kb) * FRAG_H + lane * 16);
                acc = WMMA_F16(A, B2[kb], acc);
            }
            v8h g;
#pragma unroll
            for (int i = 0; i < 8; ++i) g[i] = (half_t)fmaxf(acc[i], 0.0f);
            *(v8h*)(ldsH + ln * 72 + mt * 16 + hi * 8) = g;
        }
        asm volatile("s_wait_dscnt 0" ::: "memory");

        // ---- stage 3: O^T = relu(Wn^T @ G^T), 16x2 = 32 WMMA; LDS-transposed stores
        v16h B3[2];
        B3[0] = *(const v16h*)(ldsH + ln * 72 + hi * 16);
        B3[1] = *(const v16h*)(ldsH + ln * 72 + 32 + hi * 16);
        asm volatile("" ::: "memory");
        const half_t* wn_q = wsf + OFF_WN_H + (q * 32) * FRAG_H;
        const int dr = (q == 1 || q == 2) ? 1 : 0;
        const int dc = (q == 0 || q == 1) ? 1 : 0;
#pragma unroll
        for (int ch = 0; ch < 4; ++ch) {         // 64-feature output chunks
#pragma unroll
            for (int mtl = 0; mtl < 4; ++mtl) {
                int mt = ch * 4 + mtl;
                v8f acc = {};
#pragma unroll
                for (int kb = 0; kb < 2; ++kb) {
                    v16h A = *(const v16h*)(wn_q + (mt * 2 + kb) * FRAG_H + lane * 16);
                    acc = WMMA_F16(A, B3[kb], acc);
                }
                f32x4 lo, hv;
#pragma unroll
                for (int i = 0; i < 4; ++i) {
                    lo[i] = fmaxf(acc[i], 0.0f);
                    hv[i] = fmaxf(acc[4 + i], 0.0f);
                }
                *(f32x4*)(ldsO + ln * 68 + mtl * 16 + hi * 8)     = lo;
                *(f32x4*)(ldsO + ln * 68 + mtl * 16 + hi * 8 + 4) = hv;
            }
            asm volatile("s_wait_dscnt 0" ::: "memory");
            // read back row-major: per instr = 2 fine rows x 256B contiguous
#pragma unroll
            for (int jr = 0; jr < 8; ++jr) {
                int nl = jr * 2 + hi;
                f32x4 v = *(const f32x4*)(ldsO + nl * 68 + ln * 4);
                int p  = m0 + nl;
                int rr = p / WCg;
                int cc = p - rr * WCg;
                int fi = (2 * rr + dr) * WFg + (2 * cc + dc);
                *(f32x4*)(out + (size_t)fi * OUTD + ch * 64 + ln * 4) = v;
            }
            asm volatile("" ::: "memory");       // next chunk's LDS writes stay after reads
        }
    }
}

extern "C" void kernel_launch(void* const* d_in, const int* in_sizes, int n_in,
                              void* d_out, int out_size, void* d_ws, size_t ws_size,
                              hipStream_t stream) {
    (void)in_sizes; (void)n_in; (void)out_size; (void)ws_size;
    const float* x      = (const float*)d_in[0];
    const float* W_emb  = (const float*)d_in[1];
    const float* W_edge = (const float*)d_in[2];
    const float* W_node = (const float*)d_in[3];
    // d_in[4] (edge_idx) is unused: the fine->parent mapping is analytic.
    half_t* wsf = (half_t*)d_ws;                 // needs 229376 B

    prep_weights<<<28, 256, 0, stream>>>(W_emb, W_edge, W_node, wsf);
    upsample_kernel<<<434, 256, 0, stream>>>(x, wsf, (float*)d_out);
}
